// Block_75084618269174
// MI455X (gfx1250) — matrix operand order repaired and checked by reading, not compiled, not run on hardware
//
#include <hip/hip_runtime.h>
#include <math.h>
#include <stdint.h>

#define BB    8
#define SS    1024
#define HH    1024
#define NHEAD 16
#define DH    64
#define MLPD  4096
#define ROWS  (BB*SS)   /* 8192 */
#define SIXH  (6*HH)    /* 6144 */

typedef __bf16 bf16;
typedef __attribute__((ext_vector_type(16))) __bf16 bf16x16;
typedef __attribute__((ext_vector_type(8)))  __bf16 bf16x8;
typedef __attribute__((ext_vector_type(8)))  float  f32x8;

__device__ __forceinline__ f32x8 zero8() {
  f32x8 v;
#pragma unroll
  for (int i = 0; i < 8; ++i) v[i] = 0.f;
  return v;
}

// A-fragment (16x32 bf16 tile, M=rows). Lane r(0..15),half h: row m0+r,
// chunk0 = K[h*8 .. h*8+7], chunk1 = K[16+h*8 .. 16+h*8+7].
__device__ __forceinline__ bf16x16 load_a_frag(const bf16* base, int ld, int lane) {
  int r = lane & 15, h = (lane >> 4) & 1;
  const bf16* p = base + (size_t)r * ld + h * 8;
  bf16x8 lo = *(const bf16x8*)(p);
  bf16x8 hi = *(const bf16x8*)(p + 16);
  return __builtin_shufflevector(lo, hi, 0,1,2,3,4,5,6,7,8,9,10,11,12,13,14,15);
}

// B-fragment (32x16 bf16 tile) from row-major B^T (rows = columns of B).
// Lane r,half h: column n0+r, chunk0 = K[h*8 .. h*8+7], chunk1 = K[16+h*8 .. 16+h*8+7].
__device__ __forceinline__ bf16x16 load_b_frag(const bf16* base, int ld, int lane) {
  int r = lane & 15, h = (lane >> 4) & 1;
  const bf16* p = base + (size_t)r * ld + h * 8;
  bf16x8 lo = *(const bf16x8*)(p);
  bf16x8 hi = *(const bf16x8*)(p + 16);
  return __builtin_shufflevector(lo, hi, 0,1,2,3,4,5,6,7,8,9,10,11,12,13,14,15);
}

__device__ __forceinline__ f32x8 wmma_bf16(bf16x16 a, bf16x16 b, f32x8 c) {
  return __builtin_amdgcn_wmma_f32_16x16x32_bf16(false, a, false, b, (short)0, c, false, false);
}

// Async global -> LDS copy, 16 bytes, tracked by ASYNCcnt (CDNA5 path).
__device__ __forceinline__ void async_b128(const bf16* g, const bf16* l) {
  unsigned loff = (unsigned)(uintptr_t)l;  // low 32 bits of shared ptr = LDS offset
  asm volatile("global_load_async_to_lds_b128 %0, %1, off"
               :: "v"(loff), "v"(g) : "memory");
}

// block-wide sum over 256 threads (8 wave32), broadcast to all threads.
__device__ __forceinline__ float block_sum(float v, float* red) {
#pragma unroll
  for (int m = 16; m >= 1; m >>= 1) v += __shfl_xor(v, m, 32);
  int lane = threadIdx.x & 31, w = threadIdx.x >> 5;
  __syncthreads();
  if (lane == 0) red[w] = v;
  __syncthreads();
  if (threadIdx.x == 0) {
    float t = 0.f;
#pragma unroll
    for (int i = 0; i < 8; ++i) t += red[i];
    red[8] = t;
  }
  __syncthreads();
  return red[8];
}

// ---------------------------------------------------------------- utilities
__global__ void k_cast_transpose(const float* __restrict__ in, bf16* __restrict__ out,
                                 int K, int N) {
  int idx = blockIdx.x * 256 + threadIdx.x;
  if (idx >= K * N) return;
  int k = idx / N, n = idx - k * N;
  out[(size_t)n * K + k] = (bf16)in[idx];
}

__global__ void k_silu(const float* __restrict__ c, float* __restrict__ o, int n) {
  int i = blockIdx.x * 256 + threadIdx.x;
  if (i < n) { float x = c[i]; o[i] = x / (1.f + __expf(-x)); }
}

// cm[8,6144] = silu(c) @ w_cond   (tiny GEMM, M=8)
__global__ void k_cond_gemm(const float* __restrict__ sc, const bf16* __restrict__ wT,
                            float* __restrict__ cm) {
  int idx = blockIdx.x * 256 + threadIdx.x;
  if (idx >= BB * SIXH) return;
  int m = idx / SIXH, n = idx - m * SIXH;
  const float* a = sc + (size_t)m * HH;
  const bf16*  w = wT + (size_t)n * HH;
  float acc = 0.f;
#pragma unroll 4
  for (int k = 0; k < HH; ++k) acc += a[k] * (float)w[k];
  cm[idx] = acc;
}

__global__ void k_init_stats(float* scal, int* poscnt) {
  int i = blockIdx.x * 256 + threadIdx.x;
  if (i < 8) scal[i] = 0.f;
  if (i < MLPD) poscnt[i] = 0;
}

// LayerNorm (stats fp32, out bf16) + modulate; accumulates sum(x^2) of input.
__global__ void k_ln_mod(const float* __restrict__ X, const float* __restrict__ cm,
                         int shift_off, int scale_off,
                         bf16* __restrict__ Xmod, float* __restrict__ ssq_acc) {
  __shared__ float red[9];
  int row = blockIdx.x, b = row >> 10;
  const float* xr = X + (size_t)row * HH;
  float v[4], s = 0.f, ss = 0.f;
#pragma unroll
  for (int i = 0; i < 4; ++i) {
    v[i] = xr[threadIdx.x + 256 * i];
    s += v[i]; ss += v[i] * v[i];
  }
  float tot  = block_sum(s,  red);
  float tot2 = block_sum(ss, red);
  if (threadIdx.x == 0) atomicAdd(ssq_acc, tot2);
  float mean = tot * (1.f / HH);
  float var  = tot2 * (1.f / HH) - mean * mean;
  var = var < 0.f ? 0.f : var;
  float rsq = rsqrtf(var + 1e-6f);
  const float* cb = cm + (size_t)b * SIXH;
#pragma unroll
  for (int i = 0; i < 4; ++i) {
    int col = threadIdx.x + 256 * i;
    float ln = (float)(bf16)((v[i] - mean) * rsq);  // bf16 round like reference
    Xmod[(size_t)row * HH + col] = (bf16)(ln * cb[scale_off + col] + cb[shift_off + col]);
  }
}

// ----------------------------------------------------------- WMMA GEMM core
// Block = 64x256 tile, 8 waves each owning 32x64 (2x4 WMMA accumulators).
// A/B tiles double-buffered in LDS, filled with async global->LDS copies.
// Per tile: each thread issues 1 async b128 for A (64x32 bf16 = 4KB) and
// 4 async b128 for B (256x32 bf16 = 16KB).
__device__ __forceinline__ void issue_tile(const bf16* __restrict__ A,
                                           const bf16* __restrict__ Wt, int K,
                                           int mblk, int nblk, int k0, int tid,
                                           bf16* sA, bf16* sB) {
  int rowA = tid >> 2, qa = tid & 3;
  async_b128(A + (size_t)(mblk + rowA) * K + k0 + qa * 8, sA + rowA * 32 + qa * 8);
#pragma unroll
  for (int q = 0; q < 4; ++q)
    async_b128(Wt + (size_t)(nblk + tid) * K + k0 + q * 8, sB + tid * 32 + q * 8);
}

__device__ __forceinline__ void gemm_core_lds(const bf16* __restrict__ A,
                                              const bf16* __restrict__ Wt, int K,
                                              int mblk, int nblk, int tid, int lane,
                                              int wm, int wn,
                                              bf16* sA, bf16* sB, f32x8 acc[2][4]) {
  issue_tile(A, Wt, K, mblk, nblk, 0, tid, sA, sB);
  int buf = 0;
  for (int k0 = 0; k0 < K; k0 += 32) {
    // wait for this wave's copies of the current tile, then block barrier:
    // after it, (a) current tile is complete in LDS for everyone, and
    // (b) every wave finished computing from the *other* buffer, so it is
    // safe to start overwriting it with the next tile.
    asm volatile("s_wait_asynccnt 0" ::: "memory");
    __syncthreads();
    if (k0 + 32 < K)
      issue_tile(A, Wt, K, mblk, nblk, k0 + 32, tid,
                 sA + (buf ^ 1) * (64 * 32), sB + (buf ^ 1) * (256 * 32));
    const bf16* sAc = sA + buf * (64 * 32);
    const bf16* sBc = sB + buf * (256 * 32);
    bf16x16 a0 = load_a_frag(sAc + (size_t)(wm * 32 + 0) * 32, 32, lane);
    bf16x16 a1 = load_a_frag(sAc + (size_t)(wm * 32 + 16) * 32, 32, lane);
#pragma unroll
    for (int j = 0; j < 4; ++j) {
      bf16x16 bfr = load_b_frag(sBc + (size_t)(wn * 64 + 16 * j) * 32, 32, lane);
      acc[0][j] = wmma_bf16(a0, bfr, acc[0][j]);
      acc[1][j] = wmma_bf16(a1, bfr, acc[1][j]);
    }
    buf ^= 1;
  }
}

#define GEMM_PROLOGUE(Kdim)                                       \
  __shared__ bf16 sA[2][64 * 32];                                 \
  __shared__ bf16 sB[2][256 * 32];                                \
  int tid = threadIdx.x;                                          \
  int lane = tid & 31, w = tid >> 5;                              \
  int mblk = blockIdx.y * 64, nblk = blockIdx.x * 256;            \
  int wm = w & 1, wn = w >> 1;                                    \
  int m0 = mblk + wm * 32;                                        \
  int n0 = nblk + wn * 64;                                        \
  f32x8 acc[2][4];                                                \
  for (int i = 0; i < 2; ++i)                                     \
    for (int j = 0; j < 4; ++j) acc[i][j] = zero8();              \
  gemm_core_lds(A, Wt, (Kdim), mblk, nblk, tid, lane, wm, wn,     \
                &sA[0][0], &sB[0][0], acc);                       \
  int cn = lane & 15, half = lane >> 4;                           \
  (void)m0; (void)n0;

// QKV: scatter columns into k, q, vT (vT stored [b,h,d,s] for attention B-frags)
__global__ void k_gemm_qkv(const bf16* __restrict__ A, const bf16* __restrict__ Wt,
                           bf16* __restrict__ kb, bf16* __restrict__ qb,
                           bf16* __restrict__ vT) {
  GEMM_PROLOGUE(HH)
#pragma unroll
  for (int i = 0; i < 2; ++i)
#pragma unroll
    for (int j = 0; j < 4; ++j) {
      int n = n0 + 16 * j + cn;
#pragma unroll
      for (int r = 0; r < 8; ++r) {
        int m = m0 + 16 * i + 8 * half + r;
        float val = acc[i][j][r];
        if (n < HH) {
          kb[(size_t)m * HH + n] = (bf16)val;
        } else if (n < 2 * HH) {
          qb[(size_t)m * HH + (n - HH)] = (bf16)val;
        } else {
          int c2 = n - 2 * HH, hh = c2 >> 6, d = c2 & 63;
          int b = m >> 10, s = m & 1023;
          vT[(((size_t)b * NHEAD + hh) * DH + d) * SS + s] = (bf16)val;
        }
      }
    }
}

// attn-out projection: x_mid = x + gate_msa * (y @ Wout); accumulate ||gate*proj||^2
__global__ void k_gemm_attnout(const bf16* __restrict__ A, const bf16* __restrict__ Wt,
                               const float* __restrict__ cm, const float* __restrict__ xin,
                               float* __restrict__ xmid, float* __restrict__ scal) {
  __shared__ float red[9];
  GEMM_PROLOGUE(HH)
  float ssv = 0.f;
#pragma unroll
  for (int i = 0; i < 2; ++i)
#pragma unroll
    for (int j = 0; j < 4; ++j) {
      int n = n0 + 16 * j + cn;
#pragma unroll
      for (int r = 0; r < 8; ++r) {
        int m = m0 + 16 * i + 8 * half + r;
        int b = m >> 10;
        float gate = cm[(size_t)b * SIXH + 2 * HH + n];
        float v = gate * acc[i][j][r];
        xmid[(size_t)m * HH + n] = xin[(size_t)m * HH + n] + v;
        ssv += v * v;
      }
    }
  float tot = block_sum(ssv, red);
  if (threadIdx.x == 0) atomicAdd(&scal[1], tot);
}

// MLP1: relu stats + ||pre||^2 + gelu(tanh) -> bf16 activations
__global__ void k_gemm_mlp1(const bf16* __restrict__ A, const bf16* __restrict__ Wt,
                            bf16* __restrict__ xact, int* __restrict__ poscnt,
                            float* __restrict__ scal) {
  __shared__ float red[9];
  GEMM_PROLOGUE(HH)
  float ssv = 0.f;
#pragma unroll
  for (int i = 0; i < 2; ++i)
#pragma unroll
    for (int j = 0; j < 4; ++j) {
      int n = n0 + 16 * j + cn;
      int cnt = 0;
#pragma unroll
      for (int r = 0; r < 8; ++r) {
        int m = m0 + 16 * i + 8 * half + r;
        float pre = acc[i][j][r];
        cnt += (pre > 0.f) ? 1 : 0;
        ssv += pre * pre;
        float u = 0.7978845608028654f * (pre + 0.044715f * pre * pre * pre);
        float g = 0.5f * pre * (1.f + tanhf(u));
        xact[(size_t)m * MLPD + n] = (bf16)g;
      }
      atomicAdd(&poscnt[n], cnt);
    }
  float tot = block_sum(ssv, red);
  if (threadIdx.x == 0) atomicAdd(&scal[3], tot);
}

// MLP2: out = x_mid + gate_mlp * (act @ W2); accumulate ||gate*proj||^2
__global__ void k_gemm_mlp2(const bf16* __restrict__ A, const bf16* __restrict__ Wt,
                            const float* __restrict__ cm, const float* __restrict__ xmid,
                            float* __restrict__ out, float* __restrict__ scal) {
  __shared__ float red[9];
  GEMM_PROLOGUE(MLPD)
  float ssv = 0.f;
#pragma unroll
  for (int i = 0; i < 2; ++i)
#pragma unroll
    for (int j = 0; j < 4; ++j) {
      int n = n0 + 16 * j + cn;
#pragma unroll
      for (int r = 0; r < 8; ++r) {
        int m = m0 + 16 * i + 8 * half + r;
        int b = m >> 10;
        float gate = cm[(size_t)b * SIXH + 5 * HH + n];
        float v = gate * acc[i][j][r];
        out[(size_t)m * HH + n] = xmid[(size_t)m * HH + n] + v;
        ssv += v * v;
      }
    }
  float tot = block_sum(ssv, red);
  if (threadIdx.x == 0) atomicAdd(&scal[4], tot);
}

// --------------------------------------------------------- flash attention
// S^T = K @ Q^T (so P accumulators are lane-local A-frags for P@V), 32 kv/step.
__global__ void k_flash(const bf16* __restrict__ kb, const bf16* __restrict__ qb,
                        const bf16* __restrict__ vT, bf16* __restrict__ yb,
                        float* __restrict__ scal) {
  int lane = threadIdx.x & 31, w = threadIdx.x >> 5;
  int qchunk = blockIdx.x & 7, bh = blockIdx.x >> 3;
  int b = bh >> 4, h = bh & 15;
  int q0 = qchunk * 128 + w * 16;

  const bf16* Qb = qb + (size_t)b * SS * HH + h * DH;
  const bf16* Kb = kb + (size_t)b * SS * HH + h * DH;
  const bf16* Vt = vT + ((size_t)(b * NHEAD + h) * DH) * SS;

  // Q^T B-fragments for the two dh 32-blocks, resident for whole kernel
  bf16x16 qf0 = load_b_frag(Qb + (size_t)q0 * HH + 0,  HH, lane);
  bf16x16 qf1 = load_b_frag(Qb + (size_t)q0 * HH + 32, HH, lane);

  f32x8 yacc[4];
#pragma unroll
  for (int j = 0; j < 4; ++j) yacc[j] = zero8();
  float mrun = -1e30f, lrun = 0.f;

  for (int kt = 0; kt < SS; kt += 32) {
    if (kt + 32 < SS) {  // prefetch next K/V tiles (global_prefetch_b8)
      __builtin_prefetch(Kb + (size_t)(kt + 32 + (lane & 31)) * HH, 0, 3);
      __builtin_prefetch(Vt + (size_t)(lane & 3) * 16 * SS + kt + 32, 0, 3);
    }
    // scores (transposed): s0 covers kv kt..kt+15, s1 covers kt+16..kt+31
    f32x8 s0 = zero8(), s1 = zero8();
    bf16x16 ka, kbfr;
    ka   = load_a_frag(Kb + (size_t)kt * HH + 0,        HH, lane);
    kbfr = load_a_frag(Kb + (size_t)(kt + 16) * HH + 0, HH, lane);
    s0 = wmma_bf16(ka,   qf0, s0);
    s1 = wmma_bf16(kbfr, qf0, s1);
    ka   = load_a_frag(Kb + (size_t)kt * HH + 32,        HH, lane);
    kbfr = load_a_frag(Kb + (size_t)(kt + 16) * HH + 32, HH, lane);
    s0 = wmma_bf16(ka,   qf1, s0);
    s1 = wmma_bf16(kbfr, qf1, s1);

    // scale by 1/sqrt(dh); lane holds 16 scores for q = lane&15
    float tmax = -1e30f;
#pragma unroll
    for (int r = 0; r < 8; ++r) {
      s0[r] *= 0.125f; s1[r] *= 0.125f;
      tmax = fmaxf(tmax, fmaxf(s0[r], s1[r]));
    }
    tmax = fmaxf(tmax, __shfl_xor(tmax, 16, 32));
    float mnew = fmaxf(mrun, tmax);
    float cf = __expf(mrun - mnew);
    lrun *= cf;

    float psum = 0.f;
    bf16x16 pfrag;
#pragma unroll
    for (int r = 0; r < 8; ++r) {
      float p0 = __expf(s0[r] - mnew);
      float p1 = __expf(s1[r] - mnew);
      psum += p0 + p1;
      pfrag[r]     = (bf16)p0;   // kv = half*8 + r       (chunk0 of A-frag)
      pfrag[8 + r] = (bf16)p1;   // kv = 16 + half*8 + r  (chunk1 of A-frag)
    }
    psum += __shfl_xor(psum, 16, 32);
    lrun += psum;
    mrun = mnew;

    // rescale y accumulators: row q = r + 8*half -> fetch cf from lane q
    float cfq[8];
#pragma unroll
    for (int r = 0; r < 8; ++r) cfq[r] = __shfl(cf, r + 8 * (lane >> 4), 32);
#pragma unroll
    for (int j = 0; j < 4; ++j)
#pragma unroll
      for (int r = 0; r < 8; ++r) yacc[j][r] *= cfq[r];

    // y += P @ V   (P accumulators are already in A-frag layout, lane-local)
#pragma unroll
    for (int j = 0; j < 4; ++j) {
      bf16x16 vf = load_b_frag(Vt + (size_t)(j * 16) * SS + kt, SS, lane);
      yacc[j] = wmma_bf16(pfrag, vf, yacc[j]);
    }
  }

  // normalize rows by 1/l and store y (bf16, [b,s,h*64+d])
  float linv = 1.f / lrun;
  float lq[8];
#pragma unroll
  for (int r = 0; r < 8; ++r) lq[r] = __shfl(linv, r + 8 * (lane >> 4), 32);
  int dd = lane & 15, half = lane >> 4;
#pragma unroll
  for (int j = 0; j < 4; ++j)
#pragma unroll
    for (int r = 0; r < 8; ++r) {
      int qrow = q0 + r + 8 * half;
      int col  = h * DH + j * 16 + dd;
      yb[(size_t)(b * SS + qrow) * HH + col] = (bf16)(yacc[j][r] * lq[r]);
    }

  // mean(max_k softmax) contribution: per q, max weight = 1/l
  float val = (lane < 16) ? linv : 0.f;
#pragma unroll
  for (int m = 16; m >= 1; m >>= 1) val += __shfl_xor(val, m, 32);
  if (lane == 0) atomicAdd(&scal[5], val);
}

__global__ void k_finalize(const int* __restrict__ poscnt, const float* __restrict__ scal,
                           float* __restrict__ o) {
  __shared__ float red[9];
  float d = 0.f, z = 0.f, p = 0.f;
  for (int i = threadIdx.x; i < MLPD; i += 256) {
    int c = poscnt[i];
    float frac = (float)c * (1.f / ROWS);
    d += fabsf(frac - 0.5f);
    z += (c == 0) ? 1.f : 0.f;
    p += (c == ROWS) ? 1.f : 0.f;
  }
  d = block_sum(d, red);
  z = block_sum(z, red);
  p = block_sum(p, red);
  if (threadIdx.x == 0) {
    o[0] = d * (1.f / MLPD);
    o[1] = z * (1.f / MLPD);
    o[2] = p * (1.f / MLPD);
    o[3] = scal[5] * (1.f / (float)(BB * NHEAD * SS));
    o[4] = sqrtf(scal[1] / scal[0]);      // rms(x_attn)/rms(x)
    o[5] = sqrtf(scal[4] / scal[2]);      // rms(x_mlp)/rms(x_mid)
    o[6] = sqrtf(scal[3] / ((float)ROWS * (float)MLPD));
  }
}

// ------------------------------------------------------------------- launch
static inline char* carve(char*& p, size_t bytes) {
  char* r = p;
  p += (bytes + 255) & ~(size_t)255;
  return r;
}

extern "C" void kernel_launch(void* const* d_in, const int* in_sizes, int n_in,
                              void* d_out, int out_size, void* d_ws, size_t ws_size,
                              hipStream_t stream) {
  const float* x      = (const float*)d_in[0];
  const float* c      = (const float*)d_in[1];
  const float* w_cond = (const float*)d_in[2];
  const float* w_qkv  = (const float*)d_in[3];
  const float* w_ao   = (const float*)d_in[4];
  const float* w_m1   = (const float*)d_in[5];
  const float* w_m2   = (const float*)d_in[6];
  float* out = (float*)d_out;

  char* p = (char*)d_ws;
  bf16* wcondT = (bf16*)carve(p, (size_t)SIXH * HH * sizeof(bf16));
  bf16* wqkvT  = (bf16*)carve(p, (size_t)3 * HH * HH * sizeof(bf16));
  bf16* waoT   = (bf16*)carve(p, (size_t)HH * HH * sizeof(bf16));
  bf16* wm1T   = (bf16*)carve(p, (size_t)MLPD * HH * sizeof(bf16));
  bf16* wm2T   = (bf16*)carve(p, (size_t)HH * MLPD * sizeof(bf16));
  float* siluc = (float*)carve(p, (size_t)BB * HH * sizeof(float));
  float* cm    = (float*)carve(p, (size_t)BB * SIXH * sizeof(float));
  bf16* xmod   = (bf16*)carve(p, (size_t)ROWS * HH * sizeof(bf16));
  bf16* kbuf   = (bf16*)carve(p, (size_t)ROWS * HH * sizeof(bf16));
  bf16* qbuf   = (bf16*)carve(p, (size_t)ROWS * HH * sizeof(bf16));
  bf16* vTb    = (bf16*)carve(p, (size_t)ROWS * HH * sizeof(bf16));
  bf16* ybuf   = (bf16*)carve(p, (size_t)ROWS * HH * sizeof(bf16));
  float* xmid  = (float*)carve(p, (size_t)ROWS * HH * sizeof(float));
  bf16* xact   = (bf16*)carve(p, (size_t)ROWS * MLPD * sizeof(bf16));
  int*  poscnt = (int*)carve(p, (size_t)MLPD * sizeof(int));
  float* scal  = (float*)carve(p, 8 * sizeof(float));

  // weight cast + transpose to bf16 [N,K]
  k_cast_transpose<<<(HH * SIXH + 255) / 256, 256, 0, stream>>>(w_cond, wcondT, HH, SIXH);
  k_cast_transpose<<<(HH * 3 * HH + 255) / 256, 256, 0, stream>>>(w_qkv, wqkvT, HH, 3 * HH);
  k_cast_transpose<<<(HH * HH + 255) / 256, 256, 0, stream>>>(w_ao, waoT, HH, HH);
  k_cast_transpose<<<(HH * MLPD + 255) / 256, 256, 0, stream>>>(w_m1, wm1T, HH, MLPD);
  k_cast_transpose<<<(MLPD * HH + 255) / 256, 256, 0, stream>>>(w_m2, wm2T, MLPD, HH);

  k_init_stats<<<(MLPD + 255) / 256, 256, 0, stream>>>(scal, poscnt);

  // conditioning
  k_silu<<<(BB * HH + 255) / 256, 256, 0, stream>>>(c, siluc, BB * HH);
  k_cond_gemm<<<(BB * SIXH + 255) / 256, 256, 0, stream>>>(siluc, wcondT, cm);

  // attention branch
  k_ln_mod<<<ROWS, 256, 0, stream>>>(x, cm, 0, HH, xmod, &scal[0]);
  {
    dim3 g(3 * HH / 256, ROWS / 64);
    k_gemm_qkv<<<g, 256, 0, stream>>>(xmod, wqkvT, kbuf, qbuf, vTb);
  }
  k_flash<<<BB * NHEAD * (SS / 128), 256, 0, stream>>>(kbuf, qbuf, vTb, ybuf, scal);
  {
    dim3 g(HH / 256, ROWS / 64);
    k_gemm_attnout<<<g, 256, 0, stream>>>(ybuf, waoT, cm, x, xmid, scal);
  }

  // MLP branch
  k_ln_mod<<<ROWS, 256, 0, stream>>>(xmid, cm, 3 * HH, 4 * HH, xmod, &scal[2]);
  {
    dim3 g(MLPD / 256, ROWS / 64);
    k_gemm_mlp1<<<g, 256, 0, stream>>>(xmod, wm1T, xact, poscnt, scal);
  }
  {
    dim3 g(HH / 256, ROWS / 64);
    k_gemm_mlp2<<<g, 256, 0, stream>>>(xact, wm2T, cm, xmid, out, scal);
  }

  k_finalize<<<1, 256, 0, stream>>>(poscnt, scal, out + (size_t)ROWS * HH);
}
